// CausalSelfAttention_7112465842232
// MI455X (gfx1250) — matrix operand order repaired and checked by reading, not compiled, not run on hardware
//
#include <hip/hip_runtime.h>

typedef __attribute__((ext_vector_type(16))) __bf16 v16bf;
typedef __attribute__((ext_vector_type(8)))  float  v8f;

#define T_SEQ 2048
#define C_DIM 1024
#define H_NUM 16
#define HDIM  64
#define M_ROWS 4096   // B*T

__device__ __forceinline__ unsigned short f2bf(float f) {
  union { float f; unsigned u; } v; v.f = f;
  unsigned r = v.u + 0x7FFFu + ((v.u >> 16) & 1u);   // round-to-nearest-even
  return (unsigned short)(r >> 16);
}

__device__ __forceinline__ v8f zero8() {
  v8f z = {0.f, 0.f, 0.f, 0.f, 0.f, 0.f, 0.f, 0.f};
  return z;
}

union FragU { v16bf v; uint4 q[2]; };

// Load a 16-element bf16 fragment as two 16-byte chunks.
__device__ __forceinline__ v16bf load_frag(const unsigned short* p0, const unsigned short* p1) {
  FragU u;
  u.q[0] = *reinterpret_cast<const uint4*>(p0);
  u.q[1] = *reinterpret_cast<const uint4*>(p1);
  return u.v;
}

__device__ __forceinline__ v8f wmma_bf16(v16bf a, v16bf b, v8f c) {
  return __builtin_amdgcn_wmma_f32_16x16x32_bf16(false, a, false, b, (short)0, c, false, false);
}

// ---------------- stage 0: conversions ----------------

__global__ void k_cvt_x(const float* __restrict__ x, unsigned short* __restrict__ out, int n) {
  int i = blockIdx.x * blockDim.x + threadIdx.x;
  if (i < n) out[i] = f2bf(x[i]);
}

// W [K=1024][N=1024] f32  ->  Wt [N][K] bf16
__global__ void k_cvt_wT(const float* __restrict__ w, unsigned short* __restrict__ out) {
  int i = blockIdx.x * blockDim.x + threadIdx.x;   // over 1M
  int k = i >> 10, n = i & 1023;
  out[n * 1024 + k] = f2bf(w[i]);
}

// ---------------- stage 1: fused QKV projection GEMM ----------------
// Y[4096,1024] = Xbf @ W + b ; q,k stored [B,H,T,64] bf16, v stored transposed [B,H,64,T] bf16.

__global__ __launch_bounds__(256) void k_qkv_gemm(
    const unsigned short* __restrict__ xbf,
    const unsigned short* __restrict__ wtq,
    const unsigned short* __restrict__ wtk,
    const unsigned short* __restrict__ wtv,
    const float* __restrict__ bq, const float* __restrict__ bk, const float* __restrict__ bv,
    unsigned short* __restrict__ qd, unsigned short* __restrict__ kd, unsigned short* __restrict__ vtd)
{
  const int p = blockIdx.z;
  const unsigned short* wt   = (p == 0) ? wtq : (p == 1) ? wtk : wtv;
  const float*          bias = (p == 0) ? bq  : (p == 1) ? bk  : bv;
  unsigned short*       dqk  = (p == 0) ? qd  : kd;

  const int wid  = threadIdx.x >> 5;
  const int lane = threadIdx.x & 31;
  const int hi = lane >> 4, ln = lane & 15;
  const int mW = blockIdx.x * 128 + (wid >> 2) * 64;
  const int nW = blockIdx.y * 256 + (wid & 3) * 64;

  v8f acc[4][4];
#pragma unroll
  for (int i = 0; i < 4; i++)
#pragma unroll
    for (int j = 0; j < 4; j++) acc[i][j] = zero8();

  for (int kb = 0; kb < 1024; kb += 32) {
    v16bf a[4];
#pragma unroll
    for (int i = 0; i < 4; i++) {
      const unsigned short* pa = xbf + (size_t)(mW + 16 * i + ln) * 1024 + kb + 16 * hi;
      a[i] = load_frag(pa, pa + 8);
    }
#pragma unroll
    for (int j = 0; j < 4; j++) {
      const unsigned short* pb = wt + (size_t)(nW + 16 * j + ln) * 1024 + kb + 16 * hi;
      v16bf b = load_frag(pb, pb + 8);
#pragma unroll
      for (int i = 0; i < 4; i++) acc[i][j] = wmma_bf16(a[i], b, acc[i][j]);
    }
  }

#pragma unroll
  for (int i = 0; i < 4; i++) {
#pragma unroll
    for (int j = 0; j < 4; j++) {
      const int col = nW + 16 * j + ln;
      const float bsc = bias[col];
      const int hh = col >> 6, hd = col & 63;
#pragma unroll
      for (int r = 0; r < 8; r++) {
        const int row = mW + 16 * i + 8 * hi + r;
        const int bb = row >> 11, t = row & 2047;
        const unsigned short hv = f2bf(acc[i][j][r] + bsc);
        if (p == 2)
          vtd[((size_t)(bb * H_NUM + hh) * HDIM + hd) * T_SEQ + t] = hv;
        else
          dqk[((size_t)(bb * H_NUM + hh) * T_SEQ + t) * HDIM + hd] = hv;
      }
    }
  }
}

// ---------------- stage 2: flash attention ----------------
// One wave per 16-query tile; stream 32-key tiles with online softmax.

__global__ __launch_bounds__(128) void k_attn(
    const unsigned short* __restrict__ q, const unsigned short* __restrict__ k,
    const unsigned short* __restrict__ vT, unsigned short* __restrict__ aout)
{
  __shared__ __align__(16) unsigned short Pbuf[4][16 * 32];
  const int wid  = threadIdx.x >> 5;
  const int lane = threadIdx.x & 31;
  const int hi = lane >> 4, ln = lane & 15;
  const int bh = blockIdx.y;                    // b*H + h
  const int qb = (blockIdx.x * 4 + wid) * 16;   // query tile base row
  const float scale = 0.125f;                   // 1/sqrt(64)

  const unsigned short* qh = q  + (size_t)bh * T_SEQ * HDIM;
  const unsigned short* kh = k  + (size_t)bh * T_SEQ * HDIM;
  const unsigned short* vh = vT + (size_t)bh * HDIM * T_SEQ;

  // Q A-fragments for k-dim 0..31 and 32..63
  v16bf aq0, aq1;
  {
    const unsigned short* qr = qh + (size_t)(qb + ln) * HDIM + 16 * hi;
    aq0 = load_frag(qr, qr + 8);
    aq1 = load_frag(qr + 32, qr + 40);
  }

  v8f o0 = zero8(), o1 = zero8(), o2 = zero8(), o3 = zero8();
  float mrow[8], lrow[8];
#pragma unroll
  for (int r = 0; r < 8; r++) { mrow[r] = -1e30f; lrow[r] = 0.f; }

  for (int kt = 0; kt < qb + 16; kt += 32) {
    // S tiles: keys [kt, kt+16) and [kt+16, kt+32)
    v8f s0 = zero8(), s1 = zero8();
    {
      const unsigned short* kr0 = kh + (size_t)(kt + ln) * HDIM + 16 * hi;
      const unsigned short* kr1 = kh + (size_t)(kt + 16 + ln) * HDIM + 16 * hi;
      v16bf b;
      b = load_frag(kr0, kr0 + 8);            s0 = wmma_bf16(aq0, b, s0);
      b = load_frag(kr0 + 32, kr0 + 40);      s0 = wmma_bf16(aq1, b, s0);
      b = load_frag(kr1, kr1 + 8);            s1 = wmma_bf16(aq0, b, s1);
      b = load_frag(kr1 + 32, kr1 + 40);      s1 = wmma_bf16(aq1, b, s1);
    }

    const int row0 = qb + 8 * hi;
#pragma unroll
    for (int r = 0; r < 8; r++) {
      const int row = row0 + r;
      float x0 = s0[r] * scale; if (kt + ln > row)      x0 = -1e30f;
      float x1 = s1[r] * scale; if (kt + 16 + ln > row) x1 = -1e30f;
      // row max over the 32 columns held by this half-wave
      float tm = fmaxf(x0, x1);
      tm = fmaxf(tm, __shfl_xor(tm, 1, 32));
      tm = fmaxf(tm, __shfl_xor(tm, 2, 32));
      tm = fmaxf(tm, __shfl_xor(tm, 4, 32));
      tm = fmaxf(tm, __shfl_xor(tm, 8, 32));
      const float mn = fmaxf(mrow[r], tm);
      const float al = __expf(mrow[r] - mn);
      mrow[r] = mn;
      const float p0 = __expf(x0 - mn);
      const float p1 = __expf(x1 - mn);
      float ts = p0 + p1;
      ts += __shfl_xor(ts, 1, 32);
      ts += __shfl_xor(ts, 2, 32);
      ts += __shfl_xor(ts, 4, 32);
      ts += __shfl_xor(ts, 8, 32);
      lrow[r] = lrow[r] * al + ts;
      o0[r] *= al; o1[r] *= al; o2[r] *= al; o3[r] *= al;
      // stage P (D-layout -> LDS row-major 16x32)
      Pbuf[wid][(8 * hi + r) * 32 + ln]      = f2bf(p0);
      Pbuf[wid][(8 * hi + r) * 32 + 16 + ln] = f2bf(p1);
    }
    asm volatile("s_wait_dscnt 0" ::: "memory");

    // A-fragment of P from LDS (rows = queries, k = 32 keys)
    const unsigned short* pr = &Pbuf[wid][ln * 32 + 16 * hi];
    v16bf ap = load_frag(pr, pr + 8);

    // B-fragments of V from transposed layout [64][T]: contiguous along keys
    const unsigned short* vc = vh + (size_t)ln * T_SEQ + kt + 16 * hi;
    v16bf b;
    b = load_frag(vc, vc + 8);                               o0 = wmma_bf16(ap, b, o0);
    b = load_frag(vc + 16 * T_SEQ, vc + 16 * T_SEQ + 8);     o1 = wmma_bf16(ap, b, o1);
    b = load_frag(vc + 32 * T_SEQ, vc + 32 * T_SEQ + 8);     o2 = wmma_bf16(ap, b, o2);
    b = load_frag(vc + 48 * T_SEQ, vc + 48 * T_SEQ + 8);     o3 = wmma_bf16(ap, b, o3);
  }

  // normalize + write bf16 attn output in [B,T,C] (head-concat) layout
  const int b_ = bh >> 4, h_ = bh & 15;
#pragma unroll
  for (int r = 0; r < 8; r++) {
    const float inv = 1.0f / lrow[r];
    const int t = qb + 8 * hi + r;
    unsigned short* orow = aout + ((size_t)(b_ * T_SEQ + t)) * C_DIM + h_ * HDIM + ln;
    orow[0]  = f2bf(o0[r] * inv);
    orow[16] = f2bf(o1[r] * inv);
    orow[32] = f2bf(o2[r] * inv);
    orow[48] = f2bf(o3[r] * inv);
  }
}

// ---------------- stage 3: output projection ----------------

__global__ __launch_bounds__(256) void k_out_gemm(
    const unsigned short* __restrict__ abf, const unsigned short* __restrict__ wto,
    const float* __restrict__ bo, float* __restrict__ out)
{
  const int wid  = threadIdx.x >> 5;
  const int lane = threadIdx.x & 31;
  const int hi = lane >> 4, ln = lane & 15;
  const int mW = blockIdx.x * 128 + (wid >> 2) * 64;
  const int nW = blockIdx.y * 256 + (wid & 3) * 64;

  v8f acc[4][4];
#pragma unroll
  for (int i = 0; i < 4; i++)
#pragma unroll
    for (int j = 0; j < 4; j++) acc[i][j] = zero8();

  for (int kb = 0; kb < 1024; kb += 32) {
    v16bf a[4];
#pragma unroll
    for (int i = 0; i < 4; i++) {
      const unsigned short* pa = abf + (size_t)(mW + 16 * i + ln) * 1024 + kb + 16 * hi;
      a[i] = load_frag(pa, pa + 8);
    }
#pragma unroll
    for (int j = 0; j < 4; j++) {
      const unsigned short* pb = wto + (size_t)(nW + 16 * j + ln) * 1024 + kb + 16 * hi;
      v16bf b = load_frag(pb, pb + 8);
#pragma unroll
      for (int i = 0; i < 4; i++) acc[i][j] = wmma_bf16(a[i], b, acc[i][j]);
    }
  }

#pragma unroll
  for (int i = 0; i < 4; i++) {
#pragma unroll
    for (int j = 0; j < 4; j++) {
      const int col = nW + 16 * j + ln;
      const float bsc = bo[col];
#pragma unroll
      for (int r = 0; r < 8; r++) {
        const int row = mW + 16 * i + 8 * hi + r;
        out[(size_t)row * 1024 + col] = acc[i][j][r] + bsc;
      }
    }
  }
}

// ---------------- launch ----------------

extern "C" void kernel_launch(void* const* d_in, const int* in_sizes, int n_in,
                              void* d_out, int out_size, void* d_ws, size_t ws_size,
                              hipStream_t stream) {
  const float* x  = (const float*)d_in[0];
  // d_in[1] = mask (causality computed analytically)
  const float* Wq = (const float*)d_in[2];
  const float* bq = (const float*)d_in[3];
  const float* Wk = (const float*)d_in[4];
  const float* bk = (const float*)d_in[5];
  const float* Wv = (const float*)d_in[6];
  const float* bv = (const float*)d_in[7];
  const float* Wo = (const float*)d_in[8];
  const float* bo = (const float*)d_in[9];
  float* out = (float*)d_out;

  unsigned char* ws = (unsigned char*)d_ws;
  const size_t MB = 1024 * 1024;
  unsigned short* xbf  = (unsigned short*)(ws + 0);        // 8 MB  [4096][1024]
  unsigned short* wtq  = (unsigned short*)(ws + 8  * MB);  // 2 MB  [N][K]
  unsigned short* wtk  = (unsigned short*)(ws + 10 * MB);
  unsigned short* wtv  = (unsigned short*)(ws + 12 * MB);
  unsigned short* wto  = (unsigned short*)(ws + 14 * MB);
  unsigned short* qd   = (unsigned short*)(ws + 16 * MB);  // 8 MB [B,H,T,64]
  unsigned short* kd   = (unsigned short*)(ws + 24 * MB);  // 8 MB [B,H,T,64]
  unsigned short* vtd  = (unsigned short*)(ws + 32 * MB);  // 8 MB [B,H,64,T]
  unsigned short* attn = (unsigned short*)(ws + 40 * MB);  // 8 MB [B,T,C]

  k_cvt_x<<<(M_ROWS * C_DIM) / 256, 256, 0, stream>>>(x, xbf, M_ROWS * C_DIM);
  k_cvt_wT<<<(C_DIM * C_DIM) / 256, 256, 0, stream>>>(Wq, wtq);
  k_cvt_wT<<<(C_DIM * C_DIM) / 256, 256, 0, stream>>>(Wk, wtk);
  k_cvt_wT<<<(C_DIM * C_DIM) / 256, 256, 0, stream>>>(Wv, wtv);
  k_cvt_wT<<<(C_DIM * C_DIM) / 256, 256, 0, stream>>>(Wo, wto);

  dim3 g1(M_ROWS / 128, C_DIM / 256, 3);
  k_qkv_gemm<<<g1, 256, 0, stream>>>(xbf, wtq, wtk, wtv, bq, bk, bv, qd, kd, vtd);

  dim3 g2(T_SEQ / 64, 2 * H_NUM, 1);  // 4 query tiles per block (4 waves)
  k_attn<<<g2, 128, 0, stream>>>(qd, kd, vtd, attn);

  dim3 g3(M_ROWS / 128, C_DIM / 256, 1);
  k_out_gemm<<<g3, 256, 0, stream>>>(attn, wto, bo, out);
}
